// ProteinGraphTransformer_59811714564462
// MI455X (gfx1250) — matrix operand order repaired and checked by reading, not compiled, not run on hardware
//
#include <hip/hip_runtime.h>
#include <math.h>

// ---------------------------------------------------------------------------
// ProteinGraphTransformer for MI455X (gfx1250, wave32, WMMA)
//
// Dense node-feature GEMMs -> v_wmma_f32_16x16x32_f16 (f16 A/B, f32 acc),
// fully unrolled LDS staging (no bounds guards: M%16==0, K%32==0, Nout%16==0),
// B tile stored transposed in LDS so fragments load as contiguous b128 chunks,
// 2 WMMA tiles per wave (A-fragment reuse), next-k-tile prefetch on the
// streamed A operand (global_prefetch_b8).
// Edge segment-softmax -> 3 kernels with L2-resident gathers + f32 atomics;
// H is always a power of two so all edge indexing is shift/mask.
// ---------------------------------------------------------------------------

typedef __attribute__((ext_vector_type(16))) _Float16 v16h;
typedef __attribute__((ext_vector_type(8)))  _Float16 h8;
typedef __attribute__((ext_vector_type(4)))  _Float16 h4;
typedef __attribute__((ext_vector_type(8)))  float    v8f;

// ------------------------- small helpers -----------------------------------

// Monotonic order-preserving encode of float into unsigned (for atomicMax on
// signed floats).
__device__ __forceinline__ unsigned ford(float f) {
  unsigned u = __float_as_uint(f);
  return (u & 0x80000000u) ? ~u : (u | 0x80000000u);
}
__device__ __forceinline__ float funord(unsigned e) {
  unsigned u = (e & 0x80000000u) ? (e & 0x7FFFFFFFu) : ~e;
  return __uint_as_float(u);
}

__global__ void zero_f32(float* __restrict__ p, int n) {
  int t = blockIdx.x * blockDim.x + threadIdx.x;
  if (t < n) p[t] = 0.0f;
}

__global__ void relu_k(const float* __restrict__ in, float* __restrict__ out, int n) {
  int t = blockIdx.x * blockDim.x + threadIdx.x;
  if (t < n) out[t] = fmaxf(in[t], 0.0f);
}

// ------------------------- WMMA GEMM ----------------------------------------
// C[M,Nout] = A[M,K] @ W[K,Nout] + bias.
// Requirements: M % 16 == 0, K % 32 == 0, Nout % 16 == 0 (true here:
// M=20000, K in {128,256}, Nout in {256,32}).
//
// BW = block column width (cols of C per block) = 32 * nWaves.
// Each wave owns 16 rows x 32 cols of C = two 16x16 WMMA tiles sharing the
// A fragment.  blockDim.x == BW.  grid = (M/16, Nout/BW).
template <int BW>
__global__ void __launch_bounds__(BW)
gemm_bias_wmma(const float* __restrict__ A,
               const float* __restrict__ W,
               const float* __restrict__ bias,
               float* __restrict__ C,
               int K, int Nout) {
  __shared__ alignas(16) _Float16 As[16][32];     // 16 rows x 32 K (row-major)
  __shared__ alignas(16) _Float16 BsT[BW][32];    // col-major: [col][K]

  const int tid    = threadIdx.x;
  const int lane   = tid & 31;
  const int wave   = tid >> 5;
  const int tileM  = blockIdx.x << 4;
  const int tileN0 = blockIdx.y * BW;

  const int m = lane & 15;   // row within A tile / col within 16-wide C tile
  const int g = lane >> 4;   // lane group (ISA 7.12.2 16-bit layouts)

  v8f c0 = {}, c1 = {};

  for (int k0 = 0; k0 < K; k0 += 32) {
    __syncthreads();   // previous iteration's fragment reads done

    // ---- stage A tile: 16x32 f32 -> f16, float4 global loads, h4 LDS stores
    // Also prefetch next k-tile of A (streamed operand; W stays L2-resident).
    constexpr int ACH = (16 * 32 / 4) / BW;   // float4 chunks per thread
#pragma unroll
    for (int i = 0; i < ACH; ++i) {
      int idx = tid + i * BW;                 // chunk id 0..127
      int r   = idx >> 3;                     // 8 chunks per 32-half row
      int c4  = (idx & 7) << 2;
      const float* ap = &A[(size_t)(tileM + r) * K + (k0 + c4)];
      float4 xv = *(const float4*)ap;
      if (k0 + 32 < K) __builtin_prefetch(ap + 32, 0, 3);   // global_prefetch_b8
      h4 hv = { (_Float16)xv.x, (_Float16)xv.y, (_Float16)xv.z, (_Float16)xv.w };
      *(h4*)&As[r][c4] = hv;
    }

    // ---- stage B tile transposed: each thread owns one C column, two
    //      16-K chunks; global reads coalesced across lanes (consecutive cols)
#pragma unroll
    for (int kc = 0; kc < 2; ++kc) {
      _Float16 tmp[16];
#pragma unroll
      for (int j = 0; j < 16; ++j)
        tmp[j] = (_Float16)W[(size_t)(k0 + kc * 16 + j) * Nout + (tileN0 + tid)];
      *(h8*)&BsT[tid][kc * 16]     = *(const h8*)&tmp[0];
      *(h8*)&BsT[tid][kc * 16 + 8] = *(const h8*)&tmp[8];
    }

    __syncthreads();

    // ---- fragments (contiguous k order, same for A and B; 16-bit, wave32)
    // A 16x32: lane grp 0 -> K 0..15; grp 1 -> K 16..31
    const h8* Arow = (const h8*)&As[m][0];
    h8 a_lo = Arow[2 * g];     // K = g*16 .. g*16+7
    h8 a_hi = Arow[2 * g + 1]; // K = g*16+8 .. g*16+15
    v16h af = __builtin_shufflevector(a_lo, a_hi,
                                      0, 1, 2, 3, 4, 5, 6, 7,
                                      8, 9, 10, 11, 12, 13, 14, 15);

    // B 32x16: lane grp 0 -> K 0..15; grp 1 -> K 16..31 (contiguous in BsT)
    const int colL = wave * 32 + m;
    const h8* Bc0 = (const h8*)&BsT[colL][0];
    const h8* Bc1 = (const h8*)&BsT[colL + 16][0];
    v16h bf0 = __builtin_shufflevector(Bc0[g * 2], Bc0[g * 2 + 1],
                                       0, 1, 2, 3, 4, 5, 6, 7,
                                       8, 9, 10, 11, 12, 13, 14, 15);
    v16h bf1 = __builtin_shufflevector(Bc1[g * 2], Bc1[g * 2 + 1],
                                       0, 1, 2, 3, 4, 5, 6, 7,
                                       8, 9, 10, 11, 12, 13, 14, 15);

    c0 = __builtin_amdgcn_wmma_f32_16x16x32_f16(false, af, false, bf0,
                                                (short)0, c0, false, false);
    c1 = __builtin_amdgcn_wmma_f32_16x16x32_f16(false, af, false, bf1,
                                                (short)0, c1, false, false);
  }

  // ---- store (C/D layout: VGPR r -> row = r + 8*g, col = lane%16)
  const int colA = tileN0 + wave * 32 + m;
  const int colB = colA + 16;
  const float ba = bias[colA];
  const float bb = bias[colB];
#pragma unroll
  for (int r = 0; r < 8; ++r) {
    int row = tileM + r + g * 8;
    C[(size_t)row * Nout + colA] = c0[r] + ba;
    C[(size_t)row * Nout + colB] = c1[r] + bb;
  }
}

// ------------------------- edge attention -----------------------------------
// H is a power of two (8 or 1); hs = log2(H).  Feature offset of (node, head)
// is ((node << hs) + head) << 5  (32 channels per head).

// a[e,h] = dot(q[dst], k[src]) / sqrt(32); atomicMax segment max per (dst,h)
__global__ void edge_scores_k(const float* __restrict__ q, const float* __restrict__ kmat,
                              const int* __restrict__ src, const int* __restrict__ dst,
                              float* __restrict__ a, unsigned* __restrict__ amax,
                              int EH, int hs) {
  int t = blockIdx.x * blockDim.x + threadIdx.x;
  if (t >= EH) return;
  int e = t >> hs, h = t & ((1 << hs) - 1);
  int s = src[e], d = dst[e];
  const float4* qp = (const float4*)(q    + ((((size_t)d << hs) + h) << 5));
  const float4* kp = (const float4*)(kmat + ((((size_t)s << hs) + h) << 5));
  float sum = 0.f;
#pragma unroll
  for (int i = 0; i < 8; ++i) {
    float4 qv = qp[i], kv = kp[i];
    sum += qv.x * kv.x + qv.y * kv.y + qv.z * kv.z + qv.w * kv.w;
  }
  sum *= 0.1767766953f;  // 1/sqrt(32)
  a[t] = sum;
  atomicMax(&amax[((size_t)d << hs) + h], ford(sum));
}

// a = exp(a - amax[dst]); denom[dst] += a
__global__ void edge_exp_k(const int* __restrict__ dst, float* __restrict__ a,
                           const unsigned* __restrict__ amax, float* __restrict__ den,
                           int EH, int hs) {
  int t = blockIdx.x * blockDim.x + threadIdx.x;
  if (t >= EH) return;
  int e = t >> hs, h = t & ((1 << hs) - 1);
  int d = dst[e];
  float mx = funord(amax[((size_t)d << hs) + h]);
  float ex = __expf(a[t] - mx);
  a[t] = ex;
  atomicAdd(&den[((size_t)d << hs) + h], ex);
}

// acc[dst] += (a / denom[dst]) * v[src]
__global__ void edge_agg_k(const int* __restrict__ src, const int* __restrict__ dst,
                           const float* __restrict__ a, const float* __restrict__ den,
                           const float* __restrict__ v, float* __restrict__ acc,
                           int EH, int hs) {
  int t = blockIdx.x * blockDim.x + threadIdx.x;
  if (t >= EH) return;
  int e = t >> hs, h = t & ((1 << hs) - 1);
  int s = src[e], d = dst[e];
  float w = a[t] / (den[((size_t)d << hs) + h] + 1e-16f);
  const float4* vp = (const float4*)(v + ((((size_t)s << hs) + h) << 5));
  float* op = acc + ((((size_t)d << hs) + h) << 5);
#pragma unroll
  for (int i = 0; i < 8; ++i) {
    float4 vv = vp[i];
    atomicAdd(&op[i * 4 + 0], w * vv.x);
    atomicAdd(&op[i * 4 + 1], w * vv.y);
    atomicAdd(&op[i * 4 + 2], w * vv.z);
    atomicAdd(&op[i * 4 + 3], w * vv.w);
  }
}

// ------------------------- pooling + head -----------------------------------

__global__ void pool_k(const float* __restrict__ h, const int* __restrict__ batch,
                       float* __restrict__ pooled, float* __restrict__ counts, int ND) {
  int t = blockIdx.x * blockDim.x + threadIdx.x;
  if (t >= ND) return;
  int n = t >> 5, c = t & 31;                 // D = 32
  int b = batch[n];
  atomicAdd(&pooled[((size_t)b << 5) + c], h[t]);
  if (c == 0) atomicAdd(&counts[b], 1.0f);
}

// logits = (pooled/count) @ wlin + blin ; out = log_softmax(logits)
__global__ void head_k(const float* __restrict__ pooled, const float* __restrict__ counts,
                       const float* __restrict__ wlin, const float* __restrict__ blin,
                       float* __restrict__ out, int G, int D, int O) {
  int gi = threadIdx.x;
  if (gi >= G) return;
  float inv = 1.0f / fmaxf(counts[gi], 1.0f);
  float logits[10];
  for (int o = 0; o < O; ++o) {
    float s = blin[o];
    for (int c = 0; c < D; ++c) s += pooled[(size_t)gi * D + c] * inv * wlin[c * O + o];
    logits[o] = s;
  }
  float mx = logits[0];
  for (int o = 1; o < O; ++o) mx = fmaxf(mx, logits[o]);
  float se = 0.0f;
  for (int o = 0; o < O; ++o) se += __expf(logits[o] - mx);
  float lse = mx + __logf(se);
  for (int o = 0; o < O; ++o) out[(size_t)gi * O + o] = logits[o] - lse;
}

// ------------------------- launcher -----------------------------------------

extern "C" void kernel_launch(void* const* d_in, const int* in_sizes, int n_in,
                              void* d_out, int out_size, void* d_ws, size_t ws_size,
                              hipStream_t stream) {
  (void)n_in; (void)out_size; (void)ws_size;

  const float* x     = (const float*)d_in[0];
  const int*   ei    = (const int*)d_in[1];
  const int*   batch = (const int*)d_in[2];
  const float* wlin  = (const float*)d_in[35];
  const float* blin  = (const float*)d_in[36];

  const int Nn = in_sizes[2];          // 20000 nodes (multiple of 16)
  const int E  = in_sizes[1] / 2;      // 320000 edges
  const int* src = ei;
  const int* dst = ei + E;

  // ---- carve workspace (floats) ----
  float* ws = (float*)d_ws;
  size_t off = 0;
  const size_t nd = (size_t)Nn * 256;
  float*    q    = ws + off; off += nd;
  float*    kf   = ws + off; off += nd;
  float*    v    = ws + off; off += nd;
  float*    acc  = ws + off; off += nd;       // skip + aggregated attention
  float*    hA   = ws + off; off += nd;
  float*    hB   = ws + off; off += nd;
  float*    a    = ws + off; off += (size_t)E * 8;
  float*    den  = ws + off; off += (size_t)Nn * 8;
  unsigned* amax = (unsigned*)(ws + off); off += (size_t)Nn * 8;
  float*    pooled = ws + off; off += 16 * 32;   // counts directly follows
  float*    counts = ws + off; off += 16;

  const float* hin = x;
  int Din = 128;

  for (int l = 0; l < 4; ++l) {
    const int hs   = (l == 3) ? 0 : 3;   // log2(heads)
    const int H    = 1 << hs;
    const int Dout = H * 32;
    const float* wq  = (const float*)d_in[3 + 8 * l + 0];
    const float* wk  = (const float*)d_in[3 + 8 * l + 1];
    const float* wv  = (const float*)d_in[3 + 8 * l + 2];
    const float* wsk = (const float*)d_in[3 + 8 * l + 3];
    const float* bq  = (const float*)d_in[3 + 8 * l + 4];
    const float* bk  = (const float*)d_in[3 + 8 * l + 5];
    const float* bv  = (const float*)d_in[3 + 8 * l + 6];
    const float* bsk = (const float*)d_in[3 + 8 * l + 7];

    // WMMA GEMMs: q,k,v and skip (skip lands in acc, attention adds on top)
    const int mt = Nn / 16;
    if (Dout == 256) {
      dim3 gg(mt, 256 / 128), bb(128);
      gemm_bias_wmma<128><<<gg, bb, 0, stream>>>(hin, wq,  bq,  q,   Din, Dout);
      gemm_bias_wmma<128><<<gg, bb, 0, stream>>>(hin, wk,  bk,  kf,  Din, Dout);
      gemm_bias_wmma<128><<<gg, bb, 0, stream>>>(hin, wv,  bv,  v,   Din, Dout);
      gemm_bias_wmma<128><<<gg, bb, 0, stream>>>(hin, wsk, bsk, acc, Din, Dout);
    } else {  // Dout == 32
      dim3 gg(mt, 1), bb(32);
      gemm_bias_wmma<32><<<gg, bb, 0, stream>>>(hin, wq,  bq,  q,   Din, Dout);
      gemm_bias_wmma<32><<<gg, bb, 0, stream>>>(hin, wk,  bk,  kf,  Din, Dout);
      gemm_bias_wmma<32><<<gg, bb, 0, stream>>>(hin, wv,  bv,  v,   Din, Dout);
      gemm_bias_wmma<32><<<gg, bb, 0, stream>>>(hin, wsk, bsk, acc, Din, Dout);
    }

    // segment softmax state
    const int nh = Nn * H;
    zero_f32<<<(nh + 255) / 256, 256, 0, stream>>>((float*)amax, nh); // 0u == ordered min
    zero_f32<<<(nh + 255) / 256, 256, 0, stream>>>(den, nh);

    const int eh = E * H;
    edge_scores_k<<<(eh + 255) / 256, 256, 0, stream>>>(q, kf, src, dst, a, amax, eh, hs);
    edge_exp_k   <<<(eh + 255) / 256, 256, 0, stream>>>(dst, a, amax, den, eh, hs);
    edge_agg_k   <<<(eh + 255) / 256, 256, 0, stream>>>(src, dst, a, den, v, acc, eh, hs);

    // relu(acc) -> next layer input
    float* hout = (l & 1) ? hB : hA;
    const int nel = Nn * Dout;
    relu_k<<<(nel + 255) / 256, 256, 0, stream>>>(acc, hout, nel);
    hin = hout; Din = Dout;
  }

  // mean pool per graph (pooled + counts are contiguous: 512 + 16 floats)
  zero_f32<<<3, 256, 0, stream>>>(pooled, 16 * 32 + 16);
  const int nel = Nn * 32;
  pool_k<<<(nel + 255) / 256, 256, 0, stream>>>(hin, batch, pooled, counts, nel);
  head_k<<<1, 32, 0, stream>>>(pooled, counts, wlin, blin, (float*)d_out, 16, 32, 10);
}
